// InterViewAttention_81887846466075
// MI455X (gfx1250) — matrix operand order repaired and checked by reading, not compiled, not run on hardware
//
#include <hip/hip_runtime.h>
#include <hip/hip_bf16.h>
#include <stdint.h>

typedef __attribute__((ext_vector_type(16))) _Float16 v16h;
typedef __attribute__((ext_vector_type(8)))  float    v8f;

#define NN      12288
#define DD      256
#define GAMMA_F 0.5f
#define SLOPE   0.01f
#define KCHUNKS (NN / 32)     // 384

// ---------------------------------------------------------------------------
// Kernel 1: per-row score vectors.
//   s[i] = h_eu[i]·a1 + h_lo[i]·a3     t[j] = h_po[j]·a2
// One wave (32 lanes) per row; lane-strided, coalesced loads.
// ---------------------------------------------------------------------------
__global__ void iva_stats(const float* __restrict__ h_eu,
                          const float* __restrict__ h_po,
                          const float* __restrict__ h_lo,
                          const float* __restrict__ a1,
                          const float* __restrict__ a2,
                          const float* __restrict__ a3,
                          float* __restrict__ s_out,
                          float* __restrict__ t_out) {
    const int lane = threadIdx.x & 31;
    const int w    = threadIdx.x >> 5;
    const int i    = blockIdx.x * 8 + w;

    float s_eu = 0.f, s_lo = 0.f, s_po = 0.f;
    const size_t base = (size_t)i * DD;
#pragma unroll
    for (int k = lane; k < DD; k += 32) {
        s_eu += h_eu[base + k] * a1[k];
        s_po += h_po[base + k] * a2[k];
        s_lo += h_lo[base + k] * a3[k];
    }
#pragma unroll
    for (int off = 16; off; off >>= 1) {
        s_eu += __shfl_xor(s_eu, off, 32);
        s_lo += __shfl_xor(s_lo, off, 32);
        s_po += __shfl_xor(s_po, off, 32);
    }
    if (lane == 0) {
        s_out[i] = s_eu + s_lo;
        t_out[i] = s_po;
    }
}

// ---------------------------------------------------------------------------
// Kernel 2: pre-swizzle h_po (f32) into WMMA B-fragment order (f16), global.
// Layout: frag unit = ((kk*16 + ct)*32 + lane), 16 contiguous halves (e=0..15).
//   element (lane,e) of chunk kk, col-tile ct = h_po[kk*32 + 8*(lane>=16) + e + 8*(e>=8)]
//                                              [ct*16 + (lane&15)]
// One thread per fragment-lane: 16 strided f32 reads, one 32 B store.
// ---------------------------------------------------------------------------
__global__ void iva_bswz(const float* __restrict__ h_po,
                         _Float16* __restrict__ bs) {
    const int t    = blockIdx.x * 256 + threadIdx.x;   // 0 .. KCHUNKS*16*32-1
    const int lane = t & 31;
    const int ct   = (t >> 5) & 15;
    const int kk   = t >> 9;
    const int c     = ct * 16 + (lane & 15);
    const int jbase = kk * 32 + ((lane >> 4) * 8);

    v16h v;
#pragma unroll
    for (int e = 0; e < 16; ++e)
        v[e] = (_Float16)h_po[(size_t)(jbase + e + ((e >= 8) ? 8 : 0)) * DD + c];
    ((v16h*)bs)[t] = v;
}

// ---------------------------------------------------------------------------
// Kernel 3: per-row degree, masked row-max, softmax denominator.
// One 256-thread block per row; masked scores cached in LDS (48 KB) so adj
// is read exactly once here. Nontemporal: adj (604 MB) cannot live in L2.
// ---------------------------------------------------------------------------
__global__ void iva_rowstats(const float* __restrict__ adj,
                             const float* __restrict__ s_in,
                             const float* __restrict__ t_in,
                             float* __restrict__ rowmax,
                             float* __restrict__ rowscale) {
    __shared__ float ev[NN];          // 48 KB
    __shared__ float redS[8], redM[8], redD[8];

    const int tid  = threadIdx.x;
    const int lane = tid & 31;
    const int w    = tid >> 5;
    const int i    = blockIdx.x;
    const float si = s_in[i];
    const size_t rb = (size_t)i * NN;
    const float NEGINF = -__builtin_inff();

    float deg = 0.f, mx = NEGINF;
    for (int j = tid; j < NN; j += 256) {
        const float a = __builtin_nontemporal_load(&adj[rb + j]);
        float e = si + t_in[j];
        e = (e > 0.f) ? e : SLOPE * e;             // leaky_relu 0.01
        const float em = (a > 0.f) ? e : NEGINF;
        ev[j] = em;
        deg += a;
        mx = fmaxf(mx, em);
    }
#pragma unroll
    for (int off = 16; off; off >>= 1) {
        deg += __shfl_xor(deg, off, 32);
        mx   = fmaxf(mx, __shfl_xor(mx, off, 32));
    }
    if (lane == 0) { redS[w] = deg; redM[w] = mx; }
    __syncthreads();
    if (tid == 0) {
        float d = 0.f, m = NEGINF;
#pragma unroll
        for (int q = 0; q < 8; ++q) { d += redS[q]; m = fmaxf(m, redM[q]); }
        redS[0] = d; redM[0] = m;
    }
    __syncthreads();
    deg = redS[0];
    mx  = redM[0];

    float denom = 0.f;
    if (mx > NEGINF) {
        for (int j = tid; j < NN; j += 256)
            denom += __expf(ev[j] - mx);          // exp(-inf - mx) == 0
    }
#pragma unroll
    for (int off = 16; off; off >>= 1) denom += __shfl_xor(denom, off, 32);
    if (lane == 0) redD[w] = denom;
    __syncthreads();
    if (tid == 0) {
        float dn = 0.f;
#pragma unroll
        for (int q = 0; q < 8; ++q) dn += redD[q];
        rowmax[i]   = mx;
        rowscale[i] = (dn > 0.f) ? (GAMMA_F * deg / dn) : 0.f;
    }
}

// ---------------------------------------------------------------------------
// Kernel 4: fused att construction + GEMM  out = att @ h_po.
// Block = 256 threads (8 waves), computes 32 rows x 256 cols.
// Per K-chunk of 32:
//   - recompute two 16x32 att tiles, stored in LDS IN A-FRAGMENT ORDER
//     (branchless; packed f16x2 b32 stores; row constants live in registers)
//   - each wave: A frags = contiguous 32 B LDS reads; B frags = contiguous
//     coalesced 32 B global reads from the pre-swizzled L2-resident array
//   - 4x v_wmma_f32_16x16x32_f16 per wave, f32 accumulation.
// adj loads are nontemporal b64; next chunk prefetched (global_prefetch).
// ---------------------------------------------------------------------------
__global__ void iva_gemm(const float* __restrict__ adj,
                         const float* __restrict__ s_in,
                         const float* __restrict__ t_in,
                         const float* __restrict__ rowmax,
                         const float* __restrict__ rowscale,
                         const _Float16* __restrict__ bs,
                         float* __restrict__ out) {
    __shared__ __align__(32) _Float16 satt[2][512];   // two 16x32 frag-order tiles

    const int tid  = threadIdx.x;
    const int lane = tid & 31;
    const int w    = tid >> 5;
    const int row0 = blockIdx.x * 32;

    // ---- fill-role mapping (fixed per thread): rows (mf, mf+16), k pair kp ----
    const int mf = tid >> 4;                 // 0..15
    const int kp = (tid & 15) * 2;           // 0,2,...,30
    const int lf = ((kp & 8) ? 16 : 0) + mf;              // frag lane
    const int ef = ((kp & 16) ? 8 : 0) + (kp & 7);        // frag elem (even)
    const int dsti = lf * 8 + (ef >> 1);                  // u32 index in tile

    const int gr0 = row0 + mf;
    const int gr1 = gr0 + 16;
    const float s0 = s_in[gr0],     s1 = s_in[gr1];
    const float mx0 = rowmax[gr0],  mx1 = rowmax[gr1];
    const float rs0 = rowscale[gr0], rs1 = rowscale[gr1];
    const float* arow0 = adj + (size_t)gr0 * NN + kp;
    const float* arow1 = adj + (size_t)gr1 * NN + kp;

    // ---- WMMA-role constants ----
    const int ct0 = w * 2, ct1 = ct0 + 1;
    const v16h* bp = (const v16h*)bs;

    v8f acc00 = {}, acc01 = {}, acc10 = {}, acc11 = {};
    uint32_t* satt0_u = (uint32_t*)&satt[0][0];
    uint32_t* satt1_u = (uint32_t*)&satt[1][0];

    for (int kk = 0; kk < KCHUNKS; ++kk) {
        const int j0 = kk * 32;

        // --- branchless att pairs for both row tiles ---
        const float t0 = t_in[j0 + kp];
        const float t1 = t_in[j0 + kp + 1];
        const uint64_t u0 = __builtin_nontemporal_load((const uint64_t*)(arow0 + j0));
        const uint64_t u1 = __builtin_nontemporal_load((const uint64_t*)(arow1 + j0));
        const float a00 = __builtin_bit_cast(float, (uint32_t)u0);
        const float a01 = __builtin_bit_cast(float, (uint32_t)(u0 >> 32));
        const float a10 = __builtin_bit_cast(float, (uint32_t)u1);
        const float a11 = __builtin_bit_cast(float, (uint32_t)(u1 >> 32));

        float e00 = s0 + t0; e00 = (e00 > 0.f) ? e00 : SLOPE * e00;
        float e01 = s0 + t1; e01 = (e01 > 0.f) ? e01 : SLOPE * e01;
        float e10 = s1 + t0; e10 = (e10 > 0.f) ? e10 : SLOPE * e10;
        float e11 = s1 + t1; e11 = (e11 > 0.f) ? e11 : SLOPE * e11;

        float p00 = __expf(e00 - mx0); p00 = (a00 > 0.f) ? p00 : 0.f;
        float p01 = __expf(e01 - mx0); p01 = (a01 > 0.f) ? p01 : 0.f;
        float p10 = __expf(e10 - mx1); p10 = (a10 > 0.f) ? p10 : 0.f;
        float p11 = __expf(e11 - mx1); p11 = (a11 > 0.f) ? p11 : 0.f;

        union { _Float16 h[2]; uint32_t u; } pk0, pk1;
        pk0.h[0] = (_Float16)(rs0 * p00 + (1.0f - GAMMA_F) * a00);
        pk0.h[1] = (_Float16)(rs0 * p01 + (1.0f - GAMMA_F) * a01);
        pk1.h[0] = (_Float16)(rs1 * p10 + (1.0f - GAMMA_F) * a10);
        pk1.h[1] = (_Float16)(rs1 * p11 + (1.0f - GAMMA_F) * a11);
        satt0_u[dsti] = pk0.u;
        satt1_u[dsti] = pk1.u;

        // prefetch next adj chunk while this one is consumed
        if (kk + 1 < KCHUNKS) {
            __builtin_prefetch(arow0 + j0 + 32, 0, 1);
            __builtin_prefetch(arow1 + j0 + 32, 0, 1);
        }
        __syncthreads();

        // --- A fragments: contiguous 32 B per lane ---
        const v16h af0 = *(const v16h*)&satt[0][lane * 16];
        const v16h af1 = *(const v16h*)&satt[1][lane * 16];
        __syncthreads();     // all LDS reads done; next iter may overwrite

        // --- B fragments: coalesced 32 B per lane from swizzled global ---
        const v16h b0 = bp[((size_t)kk * 16 + ct0) * 32 + lane];
        const v16h b1 = bp[((size_t)kk * 16 + ct1) * 32 + lane];

        acc00 = __builtin_amdgcn_wmma_f32_16x16x32_f16(
            false, af0, false, b0, (short)0, acc00, false, false);
        acc01 = __builtin_amdgcn_wmma_f32_16x16x32_f16(
            false, af0, false, b1, (short)0, acc01, false, false);
        acc10 = __builtin_amdgcn_wmma_f32_16x16x32_f16(
            false, af1, false, b0, (short)0, acc10, false, false);
        acc11 = __builtin_amdgcn_wmma_f32_16x16x32_f16(
            false, af1, false, b1, (short)0, acc11, false, false);
    }

    // --- store D tiles (f32 C/D layout: VGPR r -> M = r + 8*(lane>=16)) ---
    const int mb = 8 * (lane >> 4);
    const int n0 = ct0 * 16 + (lane & 15);
    const int n1 = n0 + 16;
#pragma unroll
    for (int r = 0; r < 8; ++r) {
        const size_t ro0 = (size_t)(row0 + mb + r) * DD;
        const size_t ro1 = (size_t)(row0 + 16 + mb + r) * DD;
        out[ro0 + n0] = acc00[r];
        out[ro0 + n1] = acc01[r];
        out[ro1 + n0] = acc10[r];
        out[ro1 + n1] = acc11[r];
    }
}

// ---------------------------------------------------------------------------
extern "C" void kernel_launch(void* const* d_in, const int* in_sizes, int n_in,
                              void* d_out, int out_size, void* d_ws, size_t ws_size,
                              hipStream_t stream) {
    const float* h_eu = (const float*)d_in[0];
    const float* h_po = (const float*)d_in[1];
    const float* h_lo = (const float*)d_in[2];
    const float* adj  = (const float*)d_in[3];
    const float* a1   = (const float*)d_in[4];
    const float* a2   = (const float*)d_in[5];
    const float* a3   = (const float*)d_in[6];
    float* out = (float*)d_out;

    char* ws = (char*)d_ws;
    float*    s_v    = (float*)(ws);
    float*    t_v    = (float*)(ws + 4 * NN);
    float*    rmax   = (float*)(ws + 8 * NN);
    float*    rscale = (float*)(ws + 12 * NN);
    _Float16* bsw    = (_Float16*)(ws + 16 * NN);   // N*D*2 = 6.3 MB, frag order

    iva_stats<<<NN / 8, 256, 0, stream>>>(h_eu, h_po, h_lo, a1, a2, a3, s_v, t_v);
    iva_bswz<<<(KCHUNKS * 16 * 32) / 256, 256, 0, stream>>>(h_po, bsw);
    iva_rowstats<<<NN, 256, 0, stream>>>(adj, s_v, t_v, rmax, rscale);
    iva_gemm<<<NN / 32, 256, 0, stream>>>(adj, s_v, t_v, rmax, rscale, bsw, out);
}
